// ExperimentalPathwayModel_17789754540234
// MI455X (gfx1250) — matrix operand order repaired and checked by reading, not compiled, not run on hardware
//
#include <hip/hip_runtime.h>
#include <hip/hip_bf16.h>

// ---------------------------------------------------------------------------
// SNN with resonate-and-fire dynamics on MI455X (gfx1250).
//   base   = fused @ W_in + b_in                      (f16 WMMA, f32 acc)
//   32x:     rec = spikes @ W_rec + b_rec             (f16 WMMA; spikes exact)
//            membrane/resonance update + heaviside    (VALU, state in VGPRs)
//   out    = (sum spikes / 32) @ W_out + b_out        (f16 WMMA; counts exact)
// Batch tiled: 128 WGs x 16 rows, recurrence is row-independent -> no global
// sync. Spikes exchanged between the 16 waves of a WG through padded LDS.
// ---------------------------------------------------------------------------

typedef _Float16 v16h __attribute__((ext_vector_type(16)));
typedef _Float16 h8   __attribute__((ext_vector_type(8)));
typedef float    v8f  __attribute__((ext_vector_type(8)));

#define B_SZ 2048
#define F_SZ 768
#define H_SZ 1024
#define O_SZ 16
#define T_STEPS 32
#define SPK_STRIDE 1032   // 1024 + 8 halfs pad: row bank-offset 4, avoids 16-way conflicts

// workspace layout (bytes)
#define FUSED16_OFF 0
#define WIN_P_OFF   (B_SZ * F_SZ * 2)            // 3,145,728
#define WREC_P_OFF  (WIN_P_OFF + F_SZ * H_SZ * 2)
#define WOUT_P_OFF  (WREC_P_OFF + H_SZ * H_SZ * 2)
// total ws use: ~6.6 MB

static __device__ __forceinline__ v16h cat8(h8 lo, h8 hi) {
  return __builtin_shufflevector(lo, hi, 0,1,2,3,4,5,6,7,8,9,10,11,12,13,14,15);
}
static __device__ __forceinline__ v8f v8f_zero() {
  v8f z = {0.f,0.f,0.f,0.f,0.f,0.f,0.f,0.f};
  return z;
}

// ---------------------------------------------------------------------------
// Prep: f32->f16 convert of fused (row-major) + pack W_in / W_rec / W_out into
// WMMA B-fragment layout:  packed[((ntile*KT + ktile)*32 + lane)*16 + i]
//   = W[ ktile*32 + (i>>3)*16 + (lane>>4)*8 + (i&7) ][ ntile*16 + (lane&15) ]
// so the hot loop's per-lane B fragment is 32 contiguous bytes (2x b128).
// ---------------------------------------------------------------------------
__global__ void snn_prep_kernel(const float* __restrict__ fused,
                                const float* __restrict__ W_in,
                                const float* __restrict__ W_rec,
                                const float* __restrict__ W_out,
                                _Float16* __restrict__ fused16,
                                _Float16* __restrict__ Win_p,
                                _Float16* __restrict__ Wrec_p,
                                _Float16* __restrict__ Wout_p)
{
  int idx = blockIdx.x * blockDim.x + threadIdx.x;
  const int N_FUSED = B_SZ * F_SZ;
  const int N_WIN   = (F_SZ / 32) * (H_SZ / 16) * 512;
  const int N_WREC  = (H_SZ / 32) * (H_SZ / 16) * 512;
  const int N_WOUT  = (H_SZ / 32) * 512;

  if (idx < N_FUSED) { fused16[idx] = (_Float16)fused[idx]; return; }
  idx -= N_FUSED;
  if (idx < N_WIN) {
    int i = idx & 15, lane = (idx >> 4) & 31, t = idx >> 9;
    int kt = t % (F_SZ / 32), nt = t / (F_SZ / 32);
    Win_p[idx] = (_Float16)W_in[(kt*32 + (i>>3)*16 + (lane>>4)*8 + (i&7)) * H_SZ + nt*16 + (lane & 15)];
    return;
  }
  idx -= N_WIN;
  if (idx < N_WREC) {
    int i = idx & 15, lane = (idx >> 4) & 31, t = idx >> 9;
    int kt = t & 31, nt = t >> 5;
    Wrec_p[idx] = (_Float16)W_rec[(kt*32 + (i>>3)*16 + (lane>>4)*8 + (i&7)) * H_SZ + nt*16 + (lane & 15)];
    return;
  }
  idx -= N_WREC;
  if (idx < N_WOUT) {
    int i = idx & 15, lane = (idx >> 4) & 31, kt = idx >> 9;
    Wout_p[idx] = (_Float16)W_out[(kt*32 + (i>>3)*16 + (lane>>4)*8 + (i&7)) * O_SZ + (lane & 15)];
  }
}

// ---------------------------------------------------------------------------
// Main kernel: 128 blocks x 512 threads (16 wave32). Wave w owns H columns
// [w*64, w*64+64) of the 16-row batch tile; per-lane register state holds
// membrane/resonance/base/spike-count for its 32 owned (row,col) elements.
// ---------------------------------------------------------------------------
__global__ __launch_bounds__(512, 1) void snn_main_kernel(
    const _Float16* __restrict__ fused16,
    const _Float16* __restrict__ Win_p,
    const _Float16* __restrict__ Wrec_p,
    const _Float16* __restrict__ Wout_p,
    const float* __restrict__ b_in,
    const float* __restrict__ b_rec,
    const float* __restrict__ b_out,
    const float* __restrict__ rdec,
    const float* __restrict__ rfrq,
    float* __restrict__ out)
{
  __shared__ _Float16 spk[16 * SPK_STRIDE];   // 33,024 B

  const int tid  = threadIdx.x;
  const int wave = tid >> 5;
  const int lane = tid & 31;
  const int g    = lane >> 4;     // half-wave group (K / M split per ISA layout)
  const int ln   = lane & 15;
  const int rowbase = blockIdx.x * 16;

  // per-lane, per-owned-column constants (4 column tiles of 16)
  float dec[4], frq[4], bin[4], brc[4];
#pragma unroll
  for (int nt = 0; nt < 4; ++nt) {
    int c = wave * 64 + nt * 16 + ln;
    dec[nt] = 0.55f + 0.4f / (1.f + __expf(-rdec[c]));
    frq[nt] = 0.10f + 0.9f / (1.f + __expf(-rfrq[c]));
    bin[nt] = b_in[c];
    brc[nt] = b_rec[c];
  }

  v8f acc[4];
#pragma unroll
  for (int nt = 0; nt < 4; ++nt) acc[nt] = v8f_zero();

  // ---- Phase 1: base = fused @ W_in -------------------------------------
  // A fragment (16-bit A 16x32, ISA 7.12.2): lane row = ln, two contiguous
  // 8-half runs at K = kt*32 + g*8 and kt*32 + 16 + g*8.
  const _Float16* frow = fused16 + (size_t)(rowbase + ln) * F_SZ;
  for (int kt = 0; kt < F_SZ / 32; ++kt) {
    h8 alo = *(const h8*)(frow + kt * 32 + g * 8);
    h8 ahi = *(const h8*)(frow + kt * 32 + 16 + g * 8);
    v16h a = cat8(alo, ahi);
#pragma unroll
    for (int nt = 0; nt < 4; ++nt) {
      const _Float16* bp =
          Win_p + (size_t)(((wave * 4 + nt) * (F_SZ / 32) + kt) * 32 + lane) * 16;
      v16h b = cat8(*(const h8*)bp, *(const h8*)(bp + 8));
      acc[nt] = __builtin_amdgcn_wmma_f32_16x16x32_f16(
          false, a, false, b, (short)0, acc[nt], false, false);
    }
  }

  // D-fragment ownership: lane holds (row = g*8 + r, col = wave*64 + nt*16 + ln)
  float base_[4][8], mem[4][8], res[4][8], cnt[4][8];
#pragma unroll
  for (int nt = 0; nt < 4; ++nt)
#pragma unroll
    for (int r = 0; r < 8; ++r) {
      base_[nt][r] = acc[nt][r] + bin[nt];
      mem[nt][r] = 0.f; res[nt][r] = 0.f; cnt[nt][r] = 0.f;
    }

  auto do_update = [&](bool first) {
#pragma unroll
    for (int nt = 0; nt < 4; ++nt) {
#pragma unroll
      for (int r = 0; r < 8; ++r) {
        float rec   = first ? 0.0f : (acc[nt][r] + brc[nt]);
        float drive = base_[nt][r] + rec;
        float nres  = dec[nt] * res[nt][r] + frq[nt] * mem[nt][r];
        float nmem  = dec[nt] * mem[nt][r] + drive - nres;
        float s     = (nmem > 1.0f) ? 1.0f : 0.0f;   // heaviside(mem - THRESHOLD)
        mem[nt][r]  = nmem - s;                       // subtract spike*THRESHOLD
        res[nt][r]  = nres;
        cnt[nt][r] += s;
        spk[(g * 8 + r) * SPK_STRIDE + wave * 64 + nt * 16 + ln] = (_Float16)s;
      }
    }
  };

  do_update(true);  // t = 0: recurrent_current == 0, no matmul

  for (int t = 1; t < T_STEPS; ++t) {
    __syncthreads();  // all spikes of t-1 visible
#pragma unroll
    for (int nt = 0; nt < 4; ++nt) acc[nt] = v8f_zero();
    for (int kt = 0; kt < H_SZ / 32; ++kt) {
      const _Float16* srow = spk + ln * SPK_STRIDE + kt * 32 + g * 8;
      v16h a = cat8(*(const h8*)srow, *(const h8*)(srow + 16));  // ds_load_b128 x2
#pragma unroll
      for (int nt = 0; nt < 4; ++nt) {
        const _Float16* bp =
            Wrec_p + (size_t)(((wave * 4 + nt) * 32 + kt) * 32 + lane) * 16;
        v16h b = cat8(*(const h8*)bp, *(const h8*)(bp + 8));     // global b128 x2, L2-hot
        acc[nt] = __builtin_amdgcn_wmma_f32_16x16x32_f16(
            false, a, false, b, (short)0, acc[nt], false, false);
      }
    }
    __syncthreads();  // all reads of old spikes done before overwriting
    do_update(false);
  }

  // ---- Phase 3: output = (cnt/32) @ W_out + b_out ------------------------
  // counts <= 32 are exact in f16; scale AFTER the matmul.
#pragma unroll
  for (int nt = 0; nt < 4; ++nt)
#pragma unroll
    for (int r = 0; r < 8; ++r)
      spk[(g * 8 + r) * SPK_STRIDE + wave * 64 + nt * 16 + ln] = (_Float16)cnt[nt][r];
  __syncthreads();

  if (wave == 0) {  // wave-uniform branch; EXEC all-1s inside -> WMMA legal
    v8f o = v8f_zero();
    for (int kt = 0; kt < H_SZ / 32; ++kt) {
      const _Float16* srow = spk + ln * SPK_STRIDE + kt * 32 + g * 8;
      v16h a = cat8(*(const h8*)srow, *(const h8*)(srow + 16));
      const _Float16* bp = Wout_p + (size_t)(kt * 32 + lane) * 16;
      v16h b = cat8(*(const h8*)bp, *(const h8*)(bp + 8));
      o = __builtin_amdgcn_wmma_f32_16x16x32_f16(
          false, a, false, b, (short)0, o, false, false);
    }
    float bo = b_out[ln];
#pragma unroll
    for (int r = 0; r < 8; ++r)
      out[(size_t)(rowbase + g * 8 + r) * O_SZ + ln] =
          o[r] * (1.0f / (float)T_STEPS) + bo;
  }
}

// ---------------------------------------------------------------------------
extern "C" void kernel_launch(void* const* d_in, const int* in_sizes, int n_in,
                              void* d_out, int out_size, void* d_ws, size_t ws_size,
                              hipStream_t stream) {
  (void)in_sizes; (void)n_in; (void)out_size; (void)ws_size;
  const float* fused = (const float*)d_in[0];
  const float* W_in  = (const float*)d_in[1];
  const float* b_in  = (const float*)d_in[2];
  const float* W_rec = (const float*)d_in[3];
  const float* b_rec = (const float*)d_in[4];
  const float* W_out = (const float*)d_in[5];
  const float* b_out = (const float*)d_in[6];
  const float* rdec  = (const float*)d_in[7];
  const float* rfrq  = (const float*)d_in[8];

  char* ws = (char*)d_ws;
  _Float16* fused16 = (_Float16*)(ws + FUSED16_OFF);
  _Float16* Win_p   = (_Float16*)(ws + WIN_P_OFF);
  _Float16* Wrec_p  = (_Float16*)(ws + WREC_P_OFF);
  _Float16* Wout_p  = (_Float16*)(ws + WOUT_P_OFF);

  const int total = B_SZ * F_SZ
                  + (F_SZ / 32) * (H_SZ / 16) * 512
                  + (H_SZ / 32) * (H_SZ / 16) * 512
                  + (H_SZ / 32) * 512;
  snn_prep_kernel<<<(total + 255) / 256, 256, 0, stream>>>(
      fused, W_in, W_rec, W_out, fused16, Win_p, Wrec_p, Wout_p);

  snn_main_kernel<<<B_SZ / 16, 512, 0, stream>>>(
      fused16, Win_p, Wrec_p, Wout_p, b_in, b_rec, b_out, rdec, rfrq,
      (float*)d_out);
}
